// PPNet_26611617366660
// MI455X (gfx1250) — matrix operand order repaired and checked by reading, not compiled, not run on hardware
//
#include <hip/hip_runtime.h>
#include <hip/hip_bf16.h>

typedef __attribute__((ext_vector_type(16))) __bf16        v16bf;
typedef __attribute__((ext_vector_type(8)))  float         v8f;
typedef __attribute__((ext_vector_type(4)))  unsigned int  u32x4;

union Frag { u32x4 u[2]; v16bf bf; };

__device__ inline unsigned short f2bf(float f) {
    unsigned int u = __float_as_uint(f);
    u += 0x7fffu + ((u >> 16) & 1u);   // round-to-nearest-even
    return (unsigned short)(u >> 16);
}
__device__ inline unsigned int pack2bf(float a, float b) {
    return (unsigned int)f2bf(a) | ((unsigned int)f2bf(b) << 16);
}

// A fragment (16x32 bf16, M x K): lane = half*16 + l15; row M=l15, kbase=half*16.
// VGPRs 0-7 <- k = kbase+0..15 (32B contiguous).
// p must already include row*ldK + half*16 + ks*32.
__device__ inline v16bf load_a_frag(const unsigned short* p) {
    Frag f; f.u[0] = *(const u32x4*)p; f.u[1] = *(const u32x4*)(p + 8); return f.bf;
}
// B fragment (32x16 bf16, K x N): lane holds column N=l15, kbase=half*16.
// VGPR j <- k = kbase+2j,2j+1  => 16 consecutive bf16 (32B contiguous).
__device__ inline v16bf load_b_frag(const unsigned short* p) {
    Frag f; f.u[0] = *(const u32x4*)p; f.u[1] = *(const u32x4*)(p + 8); return f.bf;
}

#define WMMA_BF16(a, b, c) \
    __builtin_amdgcn_wmma_f32_16x16x32_bf16(false, (a), false, (b), (short)0, (c), false, false)

// ---------------------------------------------------------------- converters
__global__ __launch_bounds__(256) void cvt_bf16_kernel(const float* __restrict__ in,
                                                       unsigned short* __restrict__ out, int n) {
    int i = blockIdx.x * 256 + threadIdx.x;
    if (i < n) out[i] = f2bf(in[i]);
}

__global__ __launch_bounds__(256) void pnorm_kernel(const float* __restrict__ protos,
                                                    float* __restrict__ pn, int P) {
    int p = blockIdx.x * 256 + threadIdx.x;
    if (p < P) {
        float s = 0.f;
        for (int c = 0; c < 192; ++c) { float v = protos[p * 192 + c]; s += v * v; }
        pn[p] = s;
    }
}

// ---------------------------------------------------------------- top-K (K=144 of 196)
// rank-by-counting; ties broken by index (matches jax top_k); selected kept in
// index order == sorted(idx). tok_list[b*145+0] = cls token, +1..144 = img tokens.
__global__ __launch_bounds__(256) void topk_kernel(const float* __restrict__ attn,
                                                   int* __restrict__ tok_list) {
    int b = blockIdx.x;
    __shared__ float sv[196];
    __shared__ int   ssel[196];
    int tid = threadIdx.x;
    if (tid < 196) sv[tid] = attn[b * 196 + tid];
    __syncthreads();
    if (tid < 196) {
        float v = sv[tid];
        int rank = 0;
        for (int i = 0; i < 196; ++i) {
            float u = sv[i];
            rank += (u > v) || (u == v && i < tid);
        }
        ssel[tid] = (rank < 144) ? 1 : 0;
    }
    __syncthreads();
    if (tid < 196 && ssel[tid]) {
        int pos = 0;
        for (int i = 0; i < tid; ++i) pos += ssel[i];
        tok_list[b * 145 + 1 + pos] = b * 197 + 1 + tid;
    }
    if (tid == 0) tok_list[b * 145] = b * 197;
}

// ---------------------------------------------------------------- fused MLP
template <int K, int N>
__device__ inline void mlp_layer(const unsigned short* __restrict__ sIn,
                                 const unsigned short* __restrict__ wbf,
                                 const float* __restrict__ bias,
                                 unsigned short* __restrict__ sOut,
                                 int lane, int wave) {
    const int half = lane >> 4, l15 = lane & 15;
    const unsigned short* ap = sIn + l15 * K + half * 16;
    for (int nt = wave; nt < N / 16; nt += 8) {
        int n = nt * 16 + l15;
        const unsigned short* bp = wbf + (size_t)n * K + half * 16;
        v8f acc = {0.f, 0.f, 0.f, 0.f, 0.f, 0.f, 0.f, 0.f};
        for (int ks = 0; ks < K / 32; ++ks) {
            v16bf a = load_a_frag(ap + ks * 32);
            v16bf b = load_b_frag(bp + ks * 32);
            acc = WMMA_BF16(a, b, acc);
        }
        float bn = bias[n];
#pragma unroll
        for (int r = 0; r < 8; ++r) {
            float v = fmaxf(acc[r] + bn, 0.0f);                 // relu
            sOut[(r + half * 8) * N + n] = f2bf(v);
        }
    }
}

__device__ inline void mlp_layer4(const unsigned short* __restrict__ sIn,
                                  const unsigned short* __restrict__ wbf,
                                  const float* __restrict__ bias,
                                  float* __restrict__ fOut, int lane, int wave) {
    const int K = 192, N = 192;
    const int half = lane >> 4, l15 = lane & 15;
    const unsigned short* ap = sIn + l15 * K + half * 16;
    for (int nt = wave; nt < N / 16; nt += 8) {
        int n = nt * 16 + l15;
        const unsigned short* bp = wbf + (size_t)n * K + half * 16;
        v8f acc = {0.f, 0.f, 0.f, 0.f, 0.f, 0.f, 0.f, 0.f};
        for (int ks = 0; ks < K / 32; ++ks) {
            v16bf a = load_a_frag(ap + ks * 32);
            v16bf b = load_b_frag(bp + ks * 32);
            acc = WMMA_BF16(a, b, acc);
        }
        float bn = bias[n];
#pragma unroll
        for (int r = 0; r < 8; ++r) {
            float v = acc[r] + bn;
            fOut[(r + half * 8) * N + n] = 1.0f / (1.0f + expf(-v));  // sigmoid
        }
    }
}

// one workgroup = 16 tokens; 1160 workgroups cover 128*145 tokens exactly.
__global__ __launch_bounds__(256) void mlp_kernel(
    const float* __restrict__ x_tokens, const int* __restrict__ tok_list,
    const unsigned short* __restrict__ w1bf, const float* __restrict__ b1,
    const unsigned short* __restrict__ w2bf, const float* __restrict__ b2,
    const unsigned short* __restrict__ w3bf, const float* __restrict__ b3,
    const unsigned short* __restrict__ w4bf, const float* __restrict__ b4,
    unsigned short* __restrict__ featbf, float* __restrict__ tnorm) {
    __shared__ __align__(16) unsigned short sX[16 * 768];   // 24 KB (reused as f32 16x192)
    __shared__ __align__(16) unsigned short sH1[16 * 384];  // 12 KB
    __shared__ __align__(16) unsigned short sH2[16 * 384];  // 12 KB
    __shared__ int sTok[16];

    const int tile = blockIdx.x, tid = threadIdx.x;
    const int lane = tid & 31, wave = tid >> 5;

    if (tid < 16) sTok[tid] = tok_list[tile * 16 + tid];
    __syncthreads();

    // gather 16 token rows (768 f32) -> bf16 LDS, packed pair stores
    for (int i = tid; i < 16 * 384; i += 256) {
        int r = i / 384, cp = i % 384;
        const float* src = x_tokens + (size_t)sTok[r] * 768 + cp * 2;
        ((unsigned int*)sX)[(size_t)r * 384 + cp] = pack2bf(src[0], src[1]);
    }
    __syncthreads();

    mlp_layer<768, 384>(sX,  w1bf, b1, sH1, lane, wave); __syncthreads();
    mlp_layer<384, 384>(sH1, w2bf, b2, sH2, lane, wave); __syncthreads();
    mlp_layer<384, 192>(sH2, w3bf, b3, sH1, lane, wave); __syncthreads();
    float* fS = (float*)sX;                                // 16*192 f32 = 12 KB, fits
    mlp_layer4(sH1, w4bf, b4, fS, lane, wave); __syncthreads();

    // write bf16 features + f32 token norms
    for (int i = tid; i < 16 * 96; i += 256) {
        int r = i / 96, cp = i % 96;
        ((unsigned int*)featbf)[(size_t)(tile * 16 + r) * 96 + cp] =
            pack2bf(fS[r * 192 + cp * 2], fS[r * 192 + cp * 2 + 1]);
    }
    if (tid < 16) {
        float s = 0.f;
        for (int c = 0; c < 192; ++c) { float v = fS[tid * 192 + c]; s += v * v; }
        tnorm[tile * 16 + tid] = s;
    }
}

// ---------------------------------------------------------------- local protos (img tokens, min-d)
// grid = (16 proto groups, 128 batches); wave -> 16 protos; 9 token tiles of 16.
// a(d)=log((d+1)/(d+eps)) decreases in relu(d), so max_n a == a(relu(min_n d_raw)).
__global__ __launch_bounds__(256) void actl_kernel(
    const unsigned short* __restrict__ protobf, const float* __restrict__ pnorm,
    const unsigned short* __restrict__ featbf, const float* __restrict__ tnorm,
    float* __restrict__ act) {
    const int b = blockIdx.y;
    const int lane = threadIdx.x & 31, wave = threadIdx.x >> 5;
    const int pt = blockIdx.x * 8 + wave;
    if (pt >= 125) return;
    const int half = lane >> 4, l15 = lane & 15;
    const int p0 = pt * 16;
    const unsigned short* ap = protobf + (size_t)(p0 + l15) * 192 + half * 16;

    float pn[8], dmin[8];
#pragma unroll
    for (int r = 0; r < 8; ++r) { pn[r] = pnorm[p0 + r + half * 8]; dmin[r] = 3.4e38f; }

    for (int tt = 0; tt < 9; ++tt) {
        const int t = b * 145 + 1 + tt * 16 + l15;        // img slots 1..144
        const unsigned short* bp = featbf + (size_t)t * 192 + half * 16;
        const float tn = tnorm[t];
        v8f acc = {0.f, 0.f, 0.f, 0.f, 0.f, 0.f, 0.f, 0.f};
#pragma unroll
        for (int ks = 0; ks < 6; ++ks) {
            v16bf a = load_a_frag(ap + ks * 32);
            v16bf bb = load_b_frag(bp + ks * 32);
            acc = WMMA_BF16(a, bb, acc);
        }
#pragma unroll
        for (int r = 0; r < 8; ++r)
            dmin[r] = fminf(dmin[r], pn[r] - 2.0f * acc[r] + tn);
    }
    // min over the 16 token-lanes within each half-wave
#pragma unroll
    for (int m = 1; m <= 8; m <<= 1)
#pragma unroll
        for (int r = 0; r < 8; ++r)
            dmin[r] = fminf(dmin[r], __shfl_xor(dmin[r], m, 32));
    if (l15 == 0) {
#pragma unroll
        for (int r = 0; r < 8; ++r) {
            float d = fmaxf(dmin[r], 0.0f);
            act[(size_t)b * 2000 + p0 + r + half * 8] = logf((d + 1.0f) / (d + 1.0e-4f));
        }
    }
}

// ---------------------------------------------------------------- global protos (cls tokens)
// D tile: M = protos, N = 16 batches. grid = (16 proto groups, 8 batch tiles).
__global__ __launch_bounds__(256) void actg_kernel(
    const unsigned short* __restrict__ protobf, const float* __restrict__ pnorm,
    const unsigned short* __restrict__ featbf, const float* __restrict__ tnorm,
    float* __restrict__ act) {
    const int bt = blockIdx.y;
    const int lane = threadIdx.x & 31, wave = threadIdx.x >> 5;
    const int pt = blockIdx.x * 8 + wave;
    if (pt >= 125) return;
    const int half = lane >> 4, l15 = lane & 15;
    const int p0 = pt * 16;
    const int nb = bt * 16 + l15;                          // batch for this column
    const int t = nb * 145;                                // cls slot
    const unsigned short* ap = protobf + (size_t)(p0 + l15) * 192 + half * 16;
    const unsigned short* bp = featbf + (size_t)t * 192 + half * 16;
    const float tn = tnorm[t];

    v8f acc = {0.f, 0.f, 0.f, 0.f, 0.f, 0.f, 0.f, 0.f};
#pragma unroll
    for (int ks = 0; ks < 6; ++ks) {
        v16bf a = load_a_frag(ap + ks * 32);
        v16bf bb = load_b_frag(bp + ks * 32);
        acc = WMMA_BF16(a, bb, acc);
    }
#pragma unroll
    for (int r = 0; r < 8; ++r) {
        int p = p0 + r + half * 8;
        float d = fmaxf(pnorm[p] - 2.0f * acc[r] + tn, 0.0f);
        act[(size_t)nb * 2000 + p] = logf((d + 1.0f) / (d + 1.0e-4f));
    }
}

// ---------------------------------------------------------------- head: 0.3*g@Wg.T + 0.7*l@Wl.T
__global__ __launch_bounds__(256) void logits_kernel(
    const float* __restrict__ actg, const float* __restrict__ actl,
    const float* __restrict__ lwg, const float* __restrict__ lw,
    float* __restrict__ out) {
    int i = blockIdx.x * 256 + threadIdx.x;                // 128*200 = 25600 exactly
    int b = i / 200, c = i % 200;
    const float4* ag = (const float4*)(actg + (size_t)b * 2000);
    const float4* al = (const float4*)(actl + (size_t)b * 2000);
    const float4* wg = (const float4*)(lwg + (size_t)c * 2000);
    const float4* wl = (const float4*)(lw  + (size_t)c * 2000);
    float sg = 0.f, sl = 0.f;
    for (int k = 0; k < 500; ++k) {
        float4 a = ag[k], w = wg[k];
        sg += a.x * w.x + a.y * w.y + a.z * w.z + a.w * w.w;
        float4 a2 = al[k], w2 = wl[k];
        sl += a2.x * w2.x + a2.y * w2.y + a2.z * w2.z + a2.w * w2.w;
    }
    out[i] = 0.3f * sg + 0.7f * sl;
}

// ---------------------------------------------------------------- launcher
extern "C" void kernel_launch(void* const* d_in, const int* in_sizes, int n_in,
                              void* d_out, int out_size, void* d_ws, size_t ws_size,
                              hipStream_t stream) {
    (void)in_sizes; (void)n_in; (void)out_size; (void)ws_size;
    const float* x_tokens = (const float*)d_in[0];
    const float* attn     = (const float*)d_in[1];
    const float* w1 = (const float*)d_in[2];  const float* b1 = (const float*)d_in[3];
    const float* w2 = (const float*)d_in[4];  const float* b2 = (const float*)d_in[5];
    const float* w3 = (const float*)d_in[6];  const float* b3 = (const float*)d_in[7];
    const float* w4 = (const float*)d_in[8];  const float* b4 = (const float*)d_in[9];
    const float* protoL = (const float*)d_in[10];
    const float* protoG = (const float*)d_in[11];
    const float* lw  = (const float*)d_in[12];
    const float* lwg = (const float*)d_in[13];
    float* out = (float*)d_out;

    char* ws = (char*)d_ws;
    size_t off = 0;
    auto alloc = [&](size_t bytes) -> char* {
        char* p = ws + off;
        off = (off + bytes + 255) & ~(size_t)255;
        return p;
    };
    unsigned short* w1bf = (unsigned short*)alloc((size_t)384 * 768 * 2);
    unsigned short* w2bf = (unsigned short*)alloc((size_t)384 * 384 * 2);
    unsigned short* w3bf = (unsigned short*)alloc((size_t)192 * 384 * 2);
    unsigned short* w4bf = (unsigned short*)alloc((size_t)192 * 192 * 2);
    unsigned short* pLbf = (unsigned short*)alloc((size_t)2000 * 192 * 2);
    unsigned short* pGbf = (unsigned short*)alloc((size_t)2000 * 192 * 2);
    float* pnL   = (float*)alloc(2000 * 4);
    float* pnG   = (float*)alloc(2000 * 4);
    int*   tokl  = (int*)alloc((size_t)128 * 145 * 4);
    unsigned short* featbf = (unsigned short*)alloc((size_t)128 * 145 * 192 * 2);
    float* tnorm = (float*)alloc((size_t)128 * 145 * 4);
    float* actL  = (float*)alloc((size_t)128 * 2000 * 4);
    float* actG  = (float*)alloc((size_t)128 * 2000 * 4);   // ~12 MB total

    cvt_bf16_kernel<<<(384 * 768 + 255) / 256, 256, 0, stream>>>(w1, w1bf, 384 * 768);
    cvt_bf16_kernel<<<(384 * 384 + 255) / 256, 256, 0, stream>>>(w2, w2bf, 384 * 384);
    cvt_bf16_kernel<<<(192 * 384 + 255) / 256, 256, 0, stream>>>(w3, w3bf, 192 * 384);
    cvt_bf16_kernel<<<(192 * 192 + 255) / 256, 256, 0, stream>>>(w4, w4bf, 192 * 192);
    cvt_bf16_kernel<<<(2000 * 192 + 255) / 256, 256, 0, stream>>>(protoL, pLbf, 2000 * 192);
    cvt_bf16_kernel<<<(2000 * 192 + 255) / 256, 256, 0, stream>>>(protoG, pGbf, 2000 * 192);
    pnorm_kernel<<<8, 256, 0, stream>>>(protoL, pnL, 2000);
    pnorm_kernel<<<8, 256, 0, stream>>>(protoG, pnG, 2000);

    topk_kernel<<<128, 256, 0, stream>>>(attn, tokl);

    mlp_kernel<<<1160, 256, 0, stream>>>(x_tokens, tokl, w1bf, b1, w2bf, b2,
                                         w3bf, b3, w4bf, b4, featbf, tnorm);

    actl_kernel<<<dim3(16, 128), 256, 0, stream>>>(pLbf, pnL, featbf, tnorm, actL);
    actg_kernel<<<dim3(16, 8),   256, 0, stream>>>(pGbf, pnG, featbf, tnorm, actG);

    logits_kernel<<<100, 256, 0, stream>>>(actG, actL, lwg, lw, out);
}
